// NonLocalBlock2d_67482526155124
// MI455X (gfx1250) — matrix operand order repaired and checked by reading, not compiled, not run on hardware
//
#include <hip/hip_runtime.h>

typedef __attribute__((ext_vector_type(16))) _Float16 v16h;
typedef __attribute__((ext_vector_type(8)))  float    v8f;

#define WMMA_F32_F16(a, b, c) \
  __builtin_amdgcn_wmma_f32_16x16x32_f16(false, (a), false, (b), (short)0, (c), false, false)

// ---------------------------------------------------------------------------
// WMMA fragment loaders (wave32, 16x16x32 f16; contiguous k order k = 16*(lane/16) + s, same for A and B).
// Both loaders read 2x 8 contiguous halves per lane -> ds_load_b128 pairs.
//
// A: row-major [16 x 32] (M x K). Lanes 0-15: K 0-15 (h0-15);
//    lanes 16-31: K 16-31.
__device__ __forceinline__ v16h load_a_frag(const _Float16* __restrict__ A, int lda, int lane) {
  int m  = lane & 15;
  int kg = (lane >> 4) << 4;            // 0 or 16
  const _Float16* r = A + m * lda;
  v16h a;
#pragma unroll
  for (int h = 0; h < 8; ++h) a[h] = r[kg + h];
#pragma unroll
  for (int h = 0; h < 8; ++h) a[8 + h] = r[kg + 8 + h];
  return a;
}

// B fragment from row-major [N x K] storage (i.e. B^T in memory): lane's 16
// K-values are contiguous. Lanes 0-15 hold K 0-15, lanes 16-31 K 16-31.
__device__ __forceinline__ v16h load_bt_frag(const _Float16* __restrict__ Bt, int ldt, int lane) {
  int n  = lane & 15;
  int kb = (lane >> 4) << 4;            // 0 or 16
  const _Float16* r = Bt + n * ldt + kb;
  v16h b;
#pragma unroll
  for (int h = 0; h < 16; ++h) b[h] = r[h];
  return b;
}

// ---------------------------------------------------------------------------
// Kernel 1: weight / BN prep.
__global__ void prep_kernel(const float* __restrict__ Wt, const float* __restrict__ bt,
                            const float* __restrict__ Wp, const float* __restrict__ bp,
                            const float* __restrict__ Wg, const float* __restrict__ bg,
                            const float* __restrict__ Wz, const float* __restrict__ bz,
                            const float* __restrict__ gamma, const float* __restrict__ beta,
                            const float* __restrict__ rmean, const float* __restrict__ rvar,
                            _Float16* __restrict__ Wallh, _Float16* __restrict__ Wzh,
                            float* __restrict__ biasAll,
                            float* __restrict__ bnScale, float* __restrict__ bnShift) {
  int i = blockIdx.x * blockDim.x + threadIdx.x;
  if (i < 128 * 256) {
    Wallh[i]         = (_Float16)Wt[i];
    Wallh[32768 + i] = (_Float16)Wp[i];
    Wallh[65536 + i] = (_Float16)Wg[i];
    Wzh[i]           = (_Float16)Wz[i];
  }
  if (i < 128) { biasAll[i] = bt[i]; biasAll[128 + i] = bp[i]; biasAll[256 + i] = bg[i]; }
  if (i < 256) {
    float inv = gamma[i] * rsqrtf(rvar[i] + 1e-5f);
    bnScale[i] = inv;
    bnShift[i] = (bz[i] - rmean[i]) * inv + beta[i];
  }
}

// ---------------------------------------------------------------------------
// Kernel 2: fused projection GEMM. out[384, N] = Wall(384x256) * x[b](256xN)+bias
// Tile: 64 out-ch x 64 positions per WG (4 waves), K=256 in 8x32.
// theta -> [N][Ci], phi -> [N][Ci], g -> [Ci][N] (transposed, for flash PV).
__global__ __launch_bounds__(128)
void proj_kernel(const float* __restrict__ x, const _Float16* __restrict__ Wallh,
                 const float* __restrict__ biasAll,
                 _Float16* __restrict__ thetah, _Float16* __restrict__ phih,
                 _Float16* __restrict__ gT) {
  constexpr int XT = 40, WLD = 40;      // 80 B rows: 16B-aligned, bank-conflict-free
  __shared__ _Float16 Xs[64 * XT];      // X tile transposed: [64 n][32 k]
  __shared__ _Float16 Ws[64 * WLD];     // W tile: [64 m][32 k]
  const int t = threadIdx.x;
  const int lane = t & 31, wv = t >> 5;
  const int n0 = blockIdx.x * 64;
  const int m0 = blockIdx.y * 64;
  const int b  = blockIdx.z;

  v8f acc[4] = {};
  const float* xb = x + (size_t)b * 256 * 4096;

  for (int kt = 0; kt < 8; ++kt) {
    { // stage X transposed: thread t -> column n = t&63, k-chunk (t>>6)*16.
      int n = t & 63, kc = (t >> 6) * 16;
      const float* src = xb + (size_t)(kt * 32 + kc) * 4096 + n0 + n;
      unsigned* dst = (unsigned*)(Xs + n * XT + kc);
#pragma unroll
      for (int j = 0; j < 8; ++j) {
        union { _Float16 h[2]; unsigned u; } p;
        p.h[0] = (_Float16)src[(2 * j) * 4096];
        p.h[1] = (_Float16)src[(2 * j + 1) * 4096];
        dst[j] = p.u;
      }
    }
    { // stage W tile [64 m][32 k] (row-major, contiguous)
      int mm = t >> 1, c0 = (t & 1) * 16;
      const uint4* src = (const uint4*)(Wallh + (size_t)(m0 + mm) * 256 + kt * 32 + c0);
      uint4* dst = (uint4*)(Ws + mm * WLD + c0);
      dst[0] = src[0]; dst[1] = src[1];
    }
    __syncthreads();
    v16h a = load_a_frag(Ws + wv * 16 * WLD, WLD, lane);
    v16h bf[4];
#pragma unroll
    for (int ns = 0; ns < 4; ++ns)
      bf[ns] = load_bt_frag(Xs + (ns * 16) * XT, XT, lane);
#pragma unroll
    for (int ns = 0; ns < 4; ++ns)
      acc[ns] = WMMA_F32_F16(a, bf[ns], acc[ns]);
    __syncthreads();
  }

  const int half = lane >> 4, col = lane & 15;
#pragma unroll
  for (int ns = 0; ns < 4; ++ns) {
    int n = n0 + ns * 16 + col;
#pragma unroll
    for (int v = 0; v < 8; ++v) {
      int co = m0 + wv * 16 + half * 8 + v;
      float val = acc[ns][v] + biasAll[co];
      int p = co >> 7, ci = co & 127;
      if (p == 2) {                                  // g: [Ci][N] (transposed)
        gT[((size_t)b * 128 + ci) * 4096 + n] = (_Float16)val;
      } else {                                       // theta/phi: [N][Ci]
        _Float16* dst = (p == 0) ? thetah : phih;
        dst[((size_t)b * 4096 + n) * 128 + ci] = (_Float16)val;
      }
    }
  }
}

// ---------------------------------------------------------------------------
// Kernel 3: flash attention. WG = 64 queries (4 waves x 16 rows). Key blocks
// of 64; online softmax; the 4096x4096 score matrix is never materialized.
__global__ __launch_bounds__(128)
void flash_attn_kernel(const _Float16* __restrict__ thetah,
                       const _Float16* __restrict__ phih,   // [B][N][Ci]
                       const _Float16* __restrict__ gT,     // [B][Ci][N]
                       _Float16* __restrict__ yh) {
  constexpr int KTLD = 136, GTLD = 72, PLD = 72;
  __shared__ _Float16 Kt[64 * KTLD];      // phi tile: [64 key][128 ci]
  __shared__ _Float16 Gt[128 * GTLD];     // gT tile:  [128 c][64 key]
  __shared__ _Float16 Ps[4 * 16 * PLD];   // per-wave P scratch [16 q][64 key]
  const int t = threadIdx.x;
  const int lane = t & 31, wv = t >> 5;
  const int b  = blockIdx.y;
  const int q0 = blockIdx.x * 64 + wv * 16;

  // Q fragments resident in VGPRs for the whole kernel
  const _Float16* Qbase = thetah + ((size_t)b * 4096 + q0) * 128;
  v16h qf[4];
#pragma unroll
  for (int ks = 0; ks < 4; ++ks) qf[ks] = load_a_frag(Qbase + ks * 32, 128, lane);

  v8f oacc[8] = {};
  float mrun[8], lrun[8];
#pragma unroll
  for (int v = 0; v < 8; ++v) { mrun[v] = -1e30f; lrun[v] = 0.f; }

  const _Float16* Kg = phih + (size_t)b * 4096 * 128;  // [N][Ci]
  const _Float16* Gg = gT + (size_t)b * 128 * 4096;    // [Ci][N]
  _Float16* Pw = Ps + wv * 16 * PLD;
  const int half = lane >> 4, col = lane & 15;
  const int kr = t >> 1, kc0 = (t & 1) * 64;           // Kt staging coords

  for (int j = 0; j < 64; ++j) {
    const int k0 = j * 64;
    { // stage phi tile [64 key][128 ci]: 2 threads per key row
      const uint4* src = (const uint4*)(Kg + (size_t)(k0 + kr) * 128 + kc0);
      uint4* dst = (uint4*)(Kt + kr * KTLD + kc0);
#pragma unroll
      for (int i = 0; i < 8; ++i) dst[i] = src[i];
    }
    { // stage gT tile [128 c][64 key]: thread t -> channel row t
      const uint4* src = (const uint4*)(Gg + (size_t)t * 4096 + k0);
      uint4* dst = (uint4*)(Gt + t * GTLD);
#pragma unroll
      for (int i = 0; i < 8; ++i) dst[i] = src[i];
    }
    if (j + 1 < 64) { // prefetch next block's tiles (global_prefetch_b8)
      __builtin_prefetch(Kg + (size_t)(k0 + 64 + kr) * 128 + kc0, 0, 1);
      __builtin_prefetch(Gg + (size_t)t * 4096 + k0 + 64, 0, 1);
    }
    __syncthreads();

    // S[16q, 64k] = Q x K : ks outer so the 4 inner WMMAs hit independent
    // accumulators (no RAW chain); B-frags batched -> one dscnt wait per group.
    v8f sacc[4] = {};
#pragma unroll
    for (int ks = 0; ks < 4; ++ks) {
      v16h bf[4];
#pragma unroll
      for (int ksb = 0; ksb < 4; ++ksb)
        bf[ksb] = load_bt_frag(Kt + (ksb * 16) * KTLD + ks * 32, KTLD, lane);
#pragma unroll
      for (int ksb = 0; ksb < 4; ++ksb)
        sacc[ksb] = WMMA_F32_F16(qf[ks], bf[ksb], sacc[ksb]);
    }

    // online softmax: row r = half*8+v lives in reg v across one 16-lane half
    float corr[8];
#pragma unroll
    for (int v = 0; v < 8; ++v) {
      float rm = fmaxf(fmaxf(sacc[0][v], sacc[1][v]), fmaxf(sacc[2][v], sacc[3][v]));
#pragma unroll
      for (int mm = 1; mm < 16; mm <<= 1) rm = fmaxf(rm, __shfl_xor(rm, mm, 32));
      float mnew = fmaxf(mrun[v], rm);
      corr[v] = __expf(mrun[v] - mnew);
      mrun[v] = mnew;
      float rs = 0.f;
#pragma unroll
      for (int ksb = 0; ksb < 4; ++ksb) {
        float p = __expf(sacc[ksb][v] - mnew);
        rs += p;
        Pw[(half * 8 + v) * PLD + ksb * 16 + col] = (_Float16)p;   // wave-private
      }
#pragma unroll
      for (int mm = 1; mm < 16; mm <<= 1) rs += __shfl_xor(rs, mm, 32);
      lrun[v] = lrun[v] * corr[v] + rs;
    }
#pragma unroll
    for (int cs = 0; cs < 8; ++cs)
#pragma unroll
      for (int v = 0; v < 8; ++v) oacc[cs][v] *= corr[v];

    // O += P x G : 2 key-steps x 8 c-subtiles = 16 WMMAs, frags batched by 4.
    // (same-wave LDS ops are in-order -> no barrier needed for Pw)
#pragma unroll
    for (int kp = 0; kp < 2; ++kp) {
      v16h a = load_a_frag(Pw + kp * 32, PLD, lane);
#pragma unroll
      for (int cg = 0; cg < 2; ++cg) {
        v16h bf[4];
#pragma unroll
        for (int i = 0; i < 4; ++i)
          bf[i] = load_bt_frag(Gt + ((cg * 4 + i) * 16) * GTLD + kp * 32, GTLD, lane);
#pragma unroll
        for (int i = 0; i < 4; ++i)
          oacc[cg * 4 + i] = WMMA_F32_F16(a, bf[i], oacc[cg * 4 + i]);
      }
    }
    __syncthreads();
  }

  // normalize and store y[b][q][ci] (f16)
#pragma unroll
  for (int v = 0; v < 8; ++v) {
    float inv = 1.0f / lrun[v];
    int q = q0 + half * 8 + v;
    _Float16* dst = yh + ((size_t)b * 4096 + q) * 128;
#pragma unroll
    for (int cs = 0; cs < 8; ++cs)
      dst[cs * 16 + col] = (_Float16)(oacc[cs][v] * inv);
  }
}

// ---------------------------------------------------------------------------
// Kernel 4: z = Wz * y^T, fused BN(eval) + residual, f32 output.
__global__ __launch_bounds__(128)
void out_proj_kernel(const _Float16* __restrict__ yh, const _Float16* __restrict__ Wzh,
                     const float* __restrict__ bnScale, const float* __restrict__ bnShift,
                     const float* __restrict__ x, float* __restrict__ out) {
  constexpr int YLD = 136;
  __shared__ _Float16 Ys[64 * YLD];       // [64 n][128 ci]
  const int t = threadIdx.x;
  const int lane = t & 31, wv = t >> 5;
  const int n0 = blockIdx.x * 64;
  const int m0 = blockIdx.y * 64;
  const int b  = blockIdx.z;

  { // stage y tile once (K=128 fits entirely)
    int r = t >> 1, c0 = (t & 1) * 64;
    const uint4* src = (const uint4*)(yh + ((size_t)b * 4096 + n0 + r) * 128 + c0);
    uint4* dst = (uint4*)(Ys + r * YLD + c0);
#pragma unroll
    for (int i = 0; i < 8; ++i) dst[i] = src[i];
  }
  __syncthreads();

  v8f acc[4] = {};
  const _Float16* Wrow = Wzh + (size_t)(m0 + wv * 16) * 128;  // small, L2 resident
#pragma unroll
  for (int ks = 0; ks < 4; ++ks) {
    v16h a = load_a_frag(Wrow + ks * 32, 128, lane);
    v16h bf[4];
#pragma unroll
    for (int ns = 0; ns < 4; ++ns)
      bf[ns] = load_bt_frag(Ys + (ns * 16) * YLD + ks * 32, YLD, lane);
#pragma unroll
    for (int ns = 0; ns < 4; ++ns)
      acc[ns] = WMMA_F32_F16(a, bf[ns], acc[ns]);
  }

  const int half = lane >> 4, col = lane & 15;
#pragma unroll
  for (int ns = 0; ns < 4; ++ns) {
    int n = n0 + ns * 16 + col;
#pragma unroll
    for (int v = 0; v < 8; ++v) {
      int co = m0 + wv * 16 + half * 8 + v;
      size_t idx = ((size_t)b * 256 + co) * 4096 + n;
      out[idx] = acc[ns][v] * bnScale[co] + bnShift[co] + x[idx];
    }
  }
}

// ---------------------------------------------------------------------------
extern "C" void kernel_launch(void* const* d_in, const int* in_sizes, int n_in,
                              void* d_out, int out_size, void* d_ws, size_t ws_size,
                              hipStream_t stream) {
  const float* x     = (const float*)d_in[0];
  const float* Wt    = (const float*)d_in[1];
  const float* bt    = (const float*)d_in[2];
  const float* Wp    = (const float*)d_in[3];
  const float* bp    = (const float*)d_in[4];
  const float* Wg    = (const float*)d_in[5];
  const float* bg    = (const float*)d_in[6];
  const float* Wz    = (const float*)d_in[7];
  const float* bz    = (const float*)d_in[8];
  const float* gamma = (const float*)d_in[9];
  const float* beta  = (const float*)d_in[10];
  const float* rmean = (const float*)d_in[11];
  const float* rvar  = (const float*)d_in[12];
  float* out = (float*)d_out;

  // Workspace layout (~33.8 MB), all 256B-aligned offsets.
  char* ws = (char*)d_ws;
  _Float16* Wallh   = (_Float16*)(ws + 0);                         // 384*256 f16
  _Float16* Wzh     = (_Float16*)(ws + 196608);                    // 256*128 f16
  float*    biasAll = (float*)   (ws + 262144);                    // 384 f32
  float*    bnScale = (float*)   (ws + 263680);                    // 256 f32
  float*    bnShift = (float*)   (ws + 264704);                    // 256 f32
  _Float16* thetah  = (_Float16*)(ws + 266240);                    // [B][N][Ci]
  _Float16* phih    = (_Float16*)(ws + 266240 + 1ull * 8388608);   // [B][N][Ci]
  _Float16* gT      = (_Float16*)(ws + 266240 + 2ull * 8388608);   // [B][Ci][N]
  _Float16* yh      = (_Float16*)(ws + 266240 + 3ull * 8388608);   // [B][N][Ci]

  prep_kernel<<<128, 256, 0, stream>>>(Wt, bt, Wp, bp, Wg, bg, Wz, bz,
                                       gamma, beta, rmean, rvar,
                                       Wallh, Wzh, biasAll, bnScale, bnShift);
  proj_kernel<<<dim3(64, 6, 8), 128, 0, stream>>>(x, Wallh, biasAll, thetah, phih, gT);
  flash_attn_kernel<<<dim3(64, 8), 128, 0, stream>>>(thetah, phih, gT, yh);
  out_proj_kernel<<<dim3(64, 4, 8), 128, 0, stream>>>(yh, Wzh, bnScale, bnShift, x, out);
}
